// AsyncHybridAttention_64261300683345
// MI455X (gfx1250) — matrix operand order repaired and checked by reading, not compiled, not run on hardware
//
#include <hip/hip_runtime.h>

typedef __attribute__((ext_vector_type(16))) _Float16 v16h;
typedef __attribute__((ext_vector_type(8)))  _Float16 v8h;
typedef __attribute__((ext_vector_type(2)))  __fp16   v2fp;   // cvt_pkrtz return type
typedef __attribute__((ext_vector_type(8)))  float    v8f;
typedef __attribute__((ext_vector_type(4)))  float    v4f;
typedef __attribute__((ext_vector_type(2)))  float    v2f;

constexpr int SEQ    = 4096;
constexpr int NH     = 16;
constexpr int HD     = 64;
constexpr int QTILE  = 128;   // q rows per workgroup (8 waves x 16)
constexpr int KVTILE = 32;    // kv cols per iteration
constexpr int NWAVES = 8;

__device__ __forceinline__ v16h pack16(v8h a0, v8h a1) {
    v16h a;
#pragma unroll
    for (int i = 0; i < 8; ++i) { a[i] = a0[i]; a[8 + i] = a1[i]; }
    return a;
}

// Flash attention forward. Transposed formulation: S^T = K * Q^T so the
// softmax (kv) axis lives in-lane; O^T = V^T * P^T. Double-buffered LDS
// staging; all A operands loaded into distinct registers so the 4+4 WMMAs
// issue back-to-back instead of serializing on an LDS round-trip each.
__global__ __launch_bounds__(256)
void fa_fwd_wmma(const float* __restrict__ Q, const float* __restrict__ K,
                 const float* __restrict__ V, float* __restrict__ O)
{
    __shared__ _Float16 sK[2][KVTILE][HD];     // [buf][kv][d]  f16
    __shared__ _Float16 sV[2][HD][KVTILE];     // [buf][d][kv]  f16 (transposed)
    __shared__ float    sO[NWAVES][16][HD];    // epilogue relayout scratch

    const int tid  = threadIdx.x;
    const int wave = tid >> 5;
    const int lane = tid & 31;
    const int l16  = lane & 15;   // q column (this lane's softmax column)
    const int hi   = lane >> 4;   // half-wave select

    const int head  = blockIdx.y;
    const int qbase = blockIdx.x * QTILE + wave * 16;

    // Staging geometry: thread -> (kv row, 8-wide d chunk)
    const int skv = tid >> 3;          // 0..31
    const int sd  = (tid & 7) * 8;     // 0,8,...,56
    auto kaddr = [&](int kv0s) { return K + ((size_t)(kv0s + skv) * NH + head) * HD + sd; };
    auto vaddr = [&](int kv0s) { return V + ((size_t)(kv0s + skv) * NH + head) * HD + sd; };
    auto cvt_store = [&](int buf, v4f f0, v4f f1, v4f g0, v4f g1) {
        v8h h8;
#pragma unroll
        for (int i = 0; i < 4; ++i) { h8[i] = (_Float16)f0[i]; h8[4 + i] = (_Float16)f1[i]; }
        *(v8h*)&sK[buf][skv][sd] = h8;
#pragma unroll
        for (int i = 0; i < 4; ++i) {
            sV[buf][sd + i][skv]     = (_Float16)g0[i];
            sV[buf][sd + 4 + i][skv] = (_Float16)g1[i];
        }
    };

    // ---- Loop-invariant B operand: Q^T, pre-scaled by (1/sqrt(D))*log2(e) ----
    // B layout (32x16): lane col = l16; halves h -> K(d) = ks*32 + hi*16 + h.
    const float scale = 0.125f * 1.44269504088896f;
    v16h bq[2];
    {
        const int q = qbase + l16;
        const float* qp = Q + ((size_t)q * NH + head) * HD;
#pragma unroll
        for (int ks = 0; ks < 2; ++ks) {
            const int d0 = ks * 32 + hi * 16;
#pragma unroll
            for (int j = 0; j < 4; ++j) {
                v4f f = *(const v4f*)(qp + d0 + j * 4);
#pragma unroll
                for (int i = 0; i < 4; ++i)
                    bq[ks][j * 4 + i] = (_Float16)(f[i] * scale);
            }
        }
    }

    // ---- Prologue: stage tile 0 into buffer 0 ----
    {
        const float* kp = kaddr(0);
        const float* vp = vaddr(0);
        v4f f0 = *(const v4f*)kp, f1 = *(const v4f*)(kp + 4);
        v4f g0 = *(const v4f*)vp, g1 = *(const v4f*)(vp + 4);
        cvt_store(0, f0, f1, g0, g1);
    }
    __syncthreads();

    // ---- Per-lane online-softmax state (one q column per lane) ----
    float m = -1e30f, l = 0.0f;
    v8f acc[4] = {};   // O^T accumulators: rows d = nb*16 + hi*8 + r, col q = l16

    int p = 0;
    for (int kv0 = 0; kv0 < SEQ; kv0 += KVTILE) {
        const bool more = (kv0 + KVTILE) < SEQ;

        // ---- Issue next tile's global loads NOW (hidden behind the WMMAs) ----
        v4f nf0, nf1, ng0, ng1;
        if (more) {
            const float* kp = kaddr(kv0 + KVTILE);
            const float* vp = vaddr(kv0 + KVTILE);
            __builtin_prefetch(kp + (size_t)KVTILE * NH * HD, 0, 1);   // tile i+2
            __builtin_prefetch(vp + (size_t)KVTILE * NH * HD, 0, 1);
            nf0 = *(const v4f*)kp; nf1 = *(const v4f*)(kp + 4);
            ng0 = *(const v4f*)vp; ng1 = *(const v4f*)(vp + 4);
        }

        // ---- Load ALL K A-operands into distinct registers (8x ds_load_b128) ----
        // A layout (16x32): lane row = l16 (kv = t*16+l16);
        //   halves h -> K(d) = ks*32 + hi*16 + h
        v16h aK[4];
#pragma unroll
        for (int t = 0; t < 2; ++t) {
#pragma unroll
            for (int ks = 0; ks < 2; ++ks) {
                const _Float16* ap = &sK[p][t * 16 + l16][ks * 32 + hi * 16];
                aK[t * 2 + ks] = pack16(*(const v8h*)ap, *(const v8h*)(ap + 8));
            }
        }

        // ---- S^T tiles: 4 back-to-back WMMAs ----
        v8f s0 = {}, s1 = {};
        s0 = __builtin_amdgcn_wmma_f32_16x16x32_f16(false, aK[0], false, bq[0], (short)0, s0, false, false);
        s0 = __builtin_amdgcn_wmma_f32_16x16x32_f16(false, aK[1], false, bq[1], (short)0, s0, false, false);
        s1 = __builtin_amdgcn_wmma_f32_16x16x32_f16(false, aK[2], false, bq[0], (short)0, s1, false, false);
        s1 = __builtin_amdgcn_wmma_f32_16x16x32_f16(false, aK[3], false, bq[1], (short)0, s1, false, false);
        // Lane holds S^T[kv = hi*8+r][q=l16] (s0) and [kv = 16+hi*8+r] (s1).

        // ---- Issue V A-operand loads now; latency hides under softmax VALU ----
        // A = V^T block: lane row = d = nb*16 + l16; halves -> kv runs.
        v16h aV[4];
#pragma unroll
        for (int nb = 0; nb < 4; ++nb) {
            const _Float16* ap = &sV[p][nb * 16 + l16][hi * 16];
            aV[nb] = pack16(*(const v8h*)ap, *(const v8h*)(ap + 8));
        }

        // ---- Online softmax: kv axis in-lane + one xor-16 exchange ----
        float t = fmaxf(s0[0], s1[0]);
#pragma unroll
        for (int r = 1; r < 8; ++r) t = fmaxf(t, fmaxf(s0[r], s1[r]));
        t = fmaxf(t, __shfl_xor(t, 16));
        const float mnew = fmaxf(m, t);
        const float corr = __builtin_amdgcn_exp2f(m - mnew);   // raw v_exp_f32
        float rs = 0.0f;
#pragma unroll
        for (int r = 0; r < 8; ++r) {
            s0[r] = __builtin_amdgcn_exp2f(s0[r] - mnew);
            s1[r] = __builtin_amdgcn_exp2f(s1[r] - mnew);
            rs += s0[r] + s1[r];
        }
        rs += __shfl_xor(rs, 16);
        l = l * corr + rs;
        m = mnew;
#pragma unroll
        for (int nb = 0; nb < 4; ++nb) acc[nb] *= corr;

        // ---- Build P^T B operand in-register (pack f16 + xor-16 exchange) ----
        // Need: lanes 0-15 halves = kv 0..15; lanes 16-31 halves = kv 16..31.
        int u0[4], u1[4], rcv[4];
#pragma unroll
        for (int j = 0; j < 4; ++j) {
            v2fp p0 = __builtin_amdgcn_cvt_pkrtz(s0[2 * j], s0[2 * j + 1]);
            v2fp p1 = __builtin_amdgcn_cvt_pkrtz(s1[2 * j], s1[2 * j + 1]);
            u0[j] = __builtin_bit_cast(int, p0);
            u1[j] = __builtin_bit_cast(int, p1);
        }
#pragma unroll
        for (int j = 0; j < 4; ++j) {
            const int snd = hi ? u0[j] : u1[j];      // partner's missing half
            rcv[j] = __shfl_xor(snd, 16);
        }
        union { v16h h; int u[8]; } pb;
#pragma unroll
        for (int j = 0; j < 4; ++j) {
            pb.u[j]     = hi ? rcv[j] : u0[j];       // kv {0..7} / {16..23}
            pb.u[4 + j] = hi ? u1[j]  : rcv[j];      // kv {8..15} / {24..31}
        }

        // ---- O^T += V^T * P^T : 4 back-to-back WMMAs ----
#pragma unroll
        for (int nb = 0; nb < 4; ++nb) {
            acc[nb] = __builtin_amdgcn_wmma_f32_16x16x32_f16(
                false, aV[nb], false, pb.h, (short)0, acc[nb], false, false);
        }

        // ---- Stage next tile into the other buffer, single barrier ----
        if (more) {
            cvt_store(p ^ 1, nf0, nf1, ng0, ng1);
            __syncthreads();
        }
        p ^= 1;
    }

    // ---- Epilogue: normalize, relayout O^T -> O via per-wave LDS, store ----
    const float inv = __builtin_amdgcn_rcpf(l);
#pragma unroll
    for (int nb = 0; nb < 4; ++nb) {
#pragma unroll
        for (int r = 0; r < 8; r += 2) {
            v2f w = { acc[nb][r] * inv, acc[nb][r + 1] * inv };
            *(v2f*)&sO[wave][l16][nb * 16 + hi * 8 + r] = w;
        }
    }
    asm volatile("s_wait_dscnt 0" ::: "memory");   // same-wave store->load order
#pragma unroll
    for (int row = 0; row < 16; ++row) {
        v2f w = *(const v2f*)&sO[wave][row][lane * 2];
        *(v2f*)(O + ((size_t)(qbase + row) * NH + head) * HD + lane * 2) = w;
    }
}

extern "C" void kernel_launch(void* const* d_in, const int* in_sizes, int n_in,
                              void* d_out, int out_size, void* d_ws, size_t ws_size,
                              hipStream_t stream) {
    const float* Q = (const float*)d_in[0];
    const float* K = (const float*)d_in[1];
    const float* V = (const float*)d_in[2];
    float* O = (float*)d_out;
    dim3 grid(SEQ / QTILE, NH);
    fa_fwd_wmma<<<grid, dim3(256), 0, stream>>>(Q, K, V, O);
}
